// Decoder_59201829208525
// MI455X (gfx1250) — compile-verified, matrix operand order checked by reading, not run on hardware
//
#include <hip/hip_runtime.h>

// ---------------------------------------------------------------------------
// Decoder layer for MI455X (gfx1250): all matmuls via v_wmma_f32_16x16x32_bf16
// GEMM uses 2x2 register tiling per wave for operand reuse (~8 MACs/byte).
// ---------------------------------------------------------------------------

typedef __attribute__((ext_vector_type(16))) __bf16 v16bf;
typedef __attribute__((ext_vector_type(8)))  float  v8f;

constexpr int   B_ = 64, T_ = 200, E_ = 512, H_ = 8, D_ = 64;
constexpr int   M_ = B_ * T_;                 // 12800 rows of activations
constexpr float SCALE_ = 0.04419417382415922f; // E^-0.5

union FragBF {
    uint4          q[2];
    unsigned short u[16];
    v16bf          v;
};

__device__ __forceinline__ unsigned short f2bf(float f) {
    unsigned u = __float_as_uint(f);
    unsigned r = u + 0x7fffu + ((u >> 16) & 1u);   // round-to-nearest-even
    return (unsigned short)(r >> 16);
}

__device__ __forceinline__ v8f wmma_bf(v16bf a, v16bf b, v8f c) {
    return __builtin_amdgcn_wmma_f32_16x16x32_bf16(false, a, false, b,
                                                   (short)0, c, false, false);
}

// ---------------------------------------------------------------------------
// Weight packing: produce bf16 W^T so B-operand fragments load contiguously
// ---------------------------------------------------------------------------

// fused (H,E,D) q/k/v weights -> WT[c][e], c = m*E + h*D + d, m in {q,k,v}
__global__ void pack_qkv_kernel(const float* __restrict__ wq,
                                const float* __restrict__ wk,
                                const float* __restrict__ wv,
                                unsigned short* __restrict__ out) {
    int tid = blockIdx.x * 256 + threadIdx.x;
    if (tid >= 3 * E_ * E_) return;
    int e  = tid & (E_ - 1);
    int c  = tid >> 9;
    int m  = c >> 9;
    int c2 = c & (E_ - 1);
    int h  = c2 >> 6;
    int d  = c2 & (D_ - 1);
    const float* src = (m == 0) ? wq : (m == 1) ? wk : wv;
    out[tid] = f2bf(src[((size_t)(h * E_ + e)) * D_ + d]);
}

// [K,N] f32 -> bf16 transposed [N,K]
__global__ void pack_t_kernel(const float* __restrict__ in,
                              unsigned short* __restrict__ out, int K, int N) {
    int tid = blockIdx.x * 256 + threadIdx.x;
    if (tid >= K * N) return;
    int n = tid / K;
    int k = tid - n * K;
    out[tid] = f2bf(in[(size_t)k * N + n]);
}

__global__ void f32_to_bf16_kernel(const float* __restrict__ in,
                                   unsigned short* __restrict__ out, int n) {
    int tid = blockIdx.x * 256 + threadIdx.x;
    if (tid < n) out[tid] = f2bf(in[tid]);
}

// ---------------------------------------------------------------------------
// LayerNorm over E=512, one row per block (256 threads, 2 elems/thread)
// ---------------------------------------------------------------------------
__global__ __launch_bounds__(256)
void ln_kernel(const float* __restrict__ x, const float* __restrict__ g,
               const float* __restrict__ b, unsigned short* __restrict__ out) {
    __shared__ float red[256];
    const int row = blockIdx.x;
    const int t   = threadIdx.x;
    const float* xr = x + (size_t)row * E_;
    float x0 = xr[t], x1 = xr[t + 256];

    red[t] = x0 + x1;
    __syncthreads();
    for (int off = 128; off > 0; off >>= 1) {
        if (t < off) red[t] += red[t + off];
        __syncthreads();
    }
    float mu = red[0] * (1.0f / E_);
    __syncthreads();

    float d0 = x0 - mu, d1 = x1 - mu;
    red[t] = d0 * d0 + d1 * d1;
    __syncthreads();
    for (int off = 128; off > 0; off >>= 1) {
        if (t < off) red[t] += red[t + off];
        __syncthreads();
    }
    float inv = rsqrtf(red[0] * (1.0f / E_) + 1e-5f);

    size_t o = (size_t)row * E_;
    out[o + t]       = f2bf(g[t] * d0 * inv + b[t]);
    out[o + t + 256] = f2bf(g[t + 256] * d1 * inv + b[t + 256]);
}

// ---------------------------------------------------------------------------
// Generic bf16 GEMM, C = A[M,K] * B[K,N], B given transposed (BT[N,K]).
// Block = 8 waves, block tile 64x128. Each wave owns a 32x32 C tile (2x2
// WMMA register tile): 8 fragment loads feed 4 WMMAs -> each fragment is
// reused twice, halving L2 traffic per MAC vs one-tile-per-wave.
// K % 32 == 0, M % 64 == 0, N % 128 == 0 at all call sites (no bounds checks).
// epi: 0 -> Cf = acc ; 1 -> Cf = acc + bias[n] + resid ; 2 -> Cb = bf16(relu(acc+bias))
// ---------------------------------------------------------------------------
__global__ __launch_bounds__(256)
void gemm_bf16_kernel(const unsigned short* __restrict__ A,
                      const unsigned short* __restrict__ BT,
                      float* __restrict__ Cf, unsigned short* __restrict__ Cb,
                      const float* __restrict__ bias,
                      const float* __restrict__ resid,
                      int K, int ldc, int epi) {
    const int lane = threadIdx.x & 31;
    const int w    = threadIdx.x >> 5;
    const int wr   = w & 1;              // 2 wave-rows
    const int wc   = w >> 1;             // 4 wave-cols
    const int m0   = blockIdx.y * 64 + wr * 32;
    const int n0   = blockIdx.x * 128 + wc * 32;

    const int col = lane & 15;
    const int kh  = (lane >> 4) * 8;

    const unsigned short* arow0 = A  + (size_t)(m0 + col) * K;
    const unsigned short* arow1 = A  + (size_t)(m0 + 16 + col) * K;
    const unsigned short* brow0 = BT + (size_t)(n0 + col) * K;
    const unsigned short* brow1 = BT + (size_t)(n0 + 16 + col) * K;

    v8f acc00 = {}, acc01 = {}, acc10 = {}, acc11 = {};
    for (int k0 = 0; k0 < K; k0 += 32) {
        FragBF a0, a1, b0, b1;
        a0.q[0] = *(const uint4*)(arow0 + k0 + kh);
        a0.q[1] = *(const uint4*)(arow0 + k0 + 16 + kh);
        a1.q[0] = *(const uint4*)(arow1 + k0 + kh);
        a1.q[1] = *(const uint4*)(arow1 + k0 + 16 + kh);
        b0.q[0] = *(const uint4*)(brow0 + k0 + kh);
        b0.q[1] = *(const uint4*)(brow0 + k0 + 16 + kh);
        b1.q[0] = *(const uint4*)(brow1 + k0 + kh);
        b1.q[1] = *(const uint4*)(brow1 + k0 + 16 + kh);
        acc00 = wmma_bf(a0.v, b0.v, acc00);
        acc01 = wmma_bf(a0.v, b1.v, acc01);
        acc10 = wmma_bf(a1.v, b0.v, acc10);
        acc11 = wmma_bf(a1.v, b1.v, acc11);
    }

    const int rbase = (lane >> 4) * 8;
    for (int tm = 0; tm < 2; ++tm) {
        for (int tn = 0; tn < 2; ++tn) {
            v8f acc = (tm == 0) ? (tn == 0 ? acc00 : acc01)
                                : (tn == 0 ? acc10 : acc11);
            const int rb = m0 + tm * 16 + rbase;
            const int c  = n0 + tn * 16 + col;
            if (epi == 0) {
                for (int i = 0; i < 8; ++i)
                    Cf[(size_t)(rb + i) * ldc + c] = acc[i];
            } else if (epi == 1) {
                float bb = bias[c];
                for (int i = 0; i < 8; ++i) {
                    size_t id = (size_t)(rb + i) * ldc + c;
                    Cf[id] = acc[i] + bb + resid[id];
                }
            } else {
                float bb = bias[c];
                for (int i = 0; i < 8; ++i) {
                    float v = acc[i] + bb;
                    v = v > 0.0f ? v : 0.0f;
                    Cb[(size_t)(rb + i) * ldc + c] = f2bf(v);
                }
            }
        }
    }
}

// ---------------------------------------------------------------------------
// Split fused qkv f32 [M, 3E] into q/k bf16 [B,H,T,D] and vT bf16 [B,H,D,T],
// applying reference masking by zeroing VALUES (not scores). SCALE folded
// into q. mode 0 = self-attn (mask q & k), mode 1 = cross-attn (mask k only).
// ---------------------------------------------------------------------------
__global__ void split_qkv_kernel(const float* __restrict__ qkv,
                                 const int* __restrict__ mask,
                                 unsigned short* __restrict__ q,
                                 unsigned short* __restrict__ k,
                                 unsigned short* __restrict__ vT, int mode) {
    size_t tid = (size_t)blockIdx.x * 256 + threadIdx.x;
    if (tid >= (size_t)M_ * 1536) return;
    int    c  = (int)(tid % 1536);
    size_t bt = tid / 1536;
    int    t  = (int)(bt % T_);
    int    b  = (int)(bt / T_);
    int    m  = c >> 9;
    int    c2 = c & 511;
    int    h  = c2 >> 6;
    int    d  = c2 & 63;
    float  v  = qkv[tid];
    int    mk = mask[b * T_ + t];
    size_t bh = (size_t)(b * H_ + h);
    if (m == 0) {
        float val = v * SCALE_;
        if (mode == 0 && !mk) val = 0.0f;
        q[(bh * T_ + t) * D_ + d] = f2bf(val);
    } else if (m == 1) {
        k[(bh * T_ + t) * D_ + d] = f2bf(mk ? v : 0.0f);
    } else {
        vT[(bh * D_ + d) * T_ + t] = f2bf(v);
    }
}

// ---------------------------------------------------------------------------
// Causal attention, flash style. One wave per (b*H+h, 16-row Q tile).
// S = Q*K^T via WMMA (K rows load identically to A frags -> natural Q.K^T),
// online softmax, P bounced through LDS C->A layout, O += P*V via WMMA
// with V pre-transposed so B-frags are contiguous.
// ---------------------------------------------------------------------------
__global__ __launch_bounds__(32)
void attn_kernel(const unsigned short* __restrict__ Q,
                 const unsigned short* __restrict__ Kk,
                 const unsigned short* __restrict__ Vt,
                 unsigned short* __restrict__ O) {
    const int lane  = threadIdx.x & 31;
    const int bh    = blockIdx.x;            // b*H + h
    const int rt    = blockIdx.y;            // 0..12
    const int b     = bh >> 3;
    const int h     = bh & 7;
    const int m0    = rt * 16;
    const int col   = lane & 15;
    const int kh    = (lane >> 4) * 8;
    const int rbase = (lane >> 4) * 8;

    __shared__ unsigned short psh[16][32];

    const uint4 z4 = make_uint4(0u, 0u, 0u, 0u);

    // Q fragments for the two 32-wide d chunks (SCALE already folded in)
    FragBF qa[2];
    {
        int R = m0 + col;
        if (R < T_) {
            const unsigned short* qr = Q + ((size_t)bh * T_ + R) * D_;
            for (int c = 0; c < 2; ++c) {
                qa[c].q[0] = *(const uint4*)(qr + c * 32 + kh);
                qa[c].q[1] = *(const uint4*)(qr + c * 32 + 16 + kh);
            }
        } else {
            qa[0].q[0] = qa[0].q[1] = qa[1].q[0] = qa[1].q[1] = z4;
        }
    }

    float mrow[8], lrow[8];
    v8f   oc[4] = {};
    for (int i = 0; i < 8; ++i) { mrow[i] = -1e30f; lrow[i] = 0.0f; }

    const int jpmax = m0 >> 5;   // last 32-key block needed under causality
    for (int jp = 0; jp <= jpmax; ++jp) {
        const int s0 = jp * 32;

        // ---- scores for two 16-key subtiles ----
        v8f sc[2];
        for (int st = 0; st < 2; ++st) {
            FragBF kb0, kb1;
            int key = s0 + st * 16 + col;
            if (key < T_) {
                const unsigned short* kr = Kk + ((size_t)bh * T_ + key) * D_;
                kb0.q[0] = *(const uint4*)(kr + kh);
                kb0.q[1] = *(const uint4*)(kr + 16 + kh);
                kb1.q[0] = *(const uint4*)(kr + 32 + kh);
                kb1.q[1] = *(const uint4*)(kr + 48 + kh);
            } else {
                kb0.q[0] = kb0.q[1] = kb1.q[0] = kb1.q[1] = z4;
            }
            v8f s = {};
            s = wmma_bf(qa[0].v, kb0.v, s);
            s = wmma_bf(qa[1].v, kb1.v, s);
            sc[st] = s;
        }

        // ---- causal mask + online softmax statistics ----
        float alpha[8];
        for (int i = 0; i < 8; ++i) {
            int   R  = m0 + rbase + i;
            float v0 = (s0 + col      <= R) ? sc[0][i] : -1e30f;
            float v1 = (s0 + 16 + col <= R) ? sc[1][i] : -1e30f;
            sc[0][i] = v0;
            sc[1][i] = v1;
            float mx = fmaxf(v0, v1);
            mx = fmaxf(mx, __shfl_xor(mx, 1));
            mx = fmaxf(mx, __shfl_xor(mx, 2));
            mx = fmaxf(mx, __shfl_xor(mx, 4));
            mx = fmaxf(mx, __shfl_xor(mx, 8));
            float mn = fmaxf(mrow[i], mx);
            alpha[i] = __expf(mrow[i] - mn);
            mrow[i]  = mn;
        }

        // ---- P = exp(S - m), row sums, stage P to LDS ----
        for (int i = 0; i < 8; ++i) {
            float p0 = __expf(sc[0][i] - mrow[i]);
            float p1 = __expf(sc[1][i] - mrow[i]);
            float rs = p0 + p1;
            rs += __shfl_xor(rs, 1);
            rs += __shfl_xor(rs, 2);
            rs += __shfl_xor(rs, 4);
            rs += __shfl_xor(rs, 8);
            lrow[i] = lrow[i] * alpha[i] + rs;
            psh[rbase + i][col]      = f2bf(p0);
            psh[rbase + i][16 + col] = f2bf(p1);
        }
        for (int dc = 0; dc < 4; ++dc)
            for (int i = 0; i < 8; ++i) oc[dc][i] *= alpha[i];
        __syncthreads();

        // ---- reload P in A-fragment layout ----
        FragBF pa;
        for (int ii = 0; ii < 8; ++ii) {
            pa.u[ii]     = psh[col][kh + ii];
            pa.u[8 + ii] = psh[col][16 + kh + ii];
        }
        __syncthreads();

        // ---- O += P * V ----
        for (int dc = 0; dc < 4; ++dc) {
            FragBF vb;
            const unsigned short* vr =
                Vt + ((size_t)bh * D_ + dc * 16 + col) * T_;
            int t0 = s0 + kh;
            int t1 = s0 + 16 + kh;
            vb.q[0] = (t0 < T_) ? *(const uint4*)(vr + t0) : z4;
            vb.q[1] = (t1 < T_) ? *(const uint4*)(vr + t1) : z4;
            oc[dc] = wmma_bf(pa.v, vb.v, oc[dc]);
        }
    }

    // ---- normalize and store [B,T,E] ----
    for (int dc = 0; dc < 4; ++dc) {
        for (int i = 0; i < 8; ++i) {
            int R = m0 + rbase + i;
            if (R < T_) {
                float l  = lrow[i];
                float vv = (l > 0.0f) ? oc[dc][i] / l : 0.0f;
                O[((size_t)b * T_ + R) * E_ + h * D_ + dc * 16 + col] = f2bf(vv);
            }
        }
    }
}

// ---------------------------------------------------------------------------
// Host-side orchestration
// ---------------------------------------------------------------------------
extern "C" void kernel_launch(void* const* d_in, const int* in_sizes, int n_in,
                              void* d_out, int out_size, void* d_ws,
                              size_t ws_size, hipStream_t stream) {
    (void)in_sizes; (void)n_in; (void)out_size; (void)ws_size;

    const float* idx      = (const float*)d_in[0];
    const float* memory   = (const float*)d_in[1];
    const int*   src_mask = (const int*)d_in[2];
    const int*   pred_mask= (const int*)d_in[3];
    const float* sa_wq = (const float*)d_in[4];
    const float* sa_wk = (const float*)d_in[5];
    const float* sa_wv = (const float*)d_in[6];
    const float* sa_wo = (const float*)d_in[7];
    const float* sa_bo = (const float*)d_in[8];
    const float* ca_wq = (const float*)d_in[9];
    const float* ca_wk = (const float*)d_in[10];
    const float* ca_wv = (const float*)d_in[11];
    const float* ca_wo = (const float*)d_in[12];
    const float* ca_bo = (const float*)d_in[13];
    const float* f_w1  = (const float*)d_in[14];
    const float* f_b1  = (const float*)d_in[15];
    const float* f_w2  = (const float*)d_in[16];
    const float* f_b2  = (const float*)d_in[17];
    const float* ln1_g = (const float*)d_in[18];
    const float* ln1_b = (const float*)d_in[19];
    const float* ln2_g = (const float*)d_in[20];
    const float* ln2_b = (const float*)d_in[21];
    const float* ln3_g = (const float*)d_in[22];
    const float* ln3_b = (const float*)d_in[23];
    float* out = (float*)d_out;

    // ---- workspace carve (all offsets 256B aligned) ----
    char* p = (char*)d_ws;
    auto take = [&](size_t bytes) {
        char* r = p;
        p += (bytes + 255) & ~(size_t)255;
        return r;
    };
    unsigned short* wsaqkv = (unsigned short*)take((size_t)3 * E_ * E_ * 2);
    unsigned short* wcaqkv = (unsigned short*)take((size_t)3 * E_ * E_ * 2);
    unsigned short* wsao   = (unsigned short*)take((size_t)E_ * E_ * 2);
    unsigned short* wcao   = (unsigned short*)take((size_t)E_ * E_ * 2);
    unsigned short* wf1    = (unsigned short*)take((size_t)E_ * 4 * E_ * 2);
    unsigned short* wf2    = (unsigned short*)take((size_t)4 * E_ * E_ * 2);
    unsigned short* memb   = (unsigned short*)take((size_t)M_ * E_ * 2);
    unsigned short* hb     = (unsigned short*)take((size_t)M_ * E_ * 2);
    unsigned short* qb     = (unsigned short*)take((size_t)M_ * E_ * 2);
    unsigned short* kb     = (unsigned short*)take((size_t)M_ * E_ * 2);
    unsigned short* vtb    = (unsigned short*)take((size_t)M_ * E_ * 2);
    unsigned short* ob     = (unsigned short*)take((size_t)M_ * E_ * 2);
    float*          x1f    = (float*)take((size_t)M_ * E_ * 4);
    float*          x2f    = (float*)take((size_t)M_ * E_ * 4);
    float*          qkvf   = (float*)take((size_t)M_ * 3 * E_ * 4);
    unsigned short* midb   = (unsigned short*)qkvf;   // alias: dead by FFN time

    auto gemm = [&](const unsigned short* A, const unsigned short* BT,
                    float* Cf, unsigned short* Cb, const float* bias,
                    const float* resid, int N, int K, int ldc, int epi) {
        dim3 g(N / 128, M_ / 64);
        gemm_bf16_kernel<<<g, 256, 0, stream>>>(A, BT, Cf, Cb, bias, resid,
                                                K, ldc, epi);
    };

    // ---- weight packing (bf16, transposed) ----
    pack_qkv_kernel<<<(3 * E_ * E_) / 256, 256, 0, stream>>>(sa_wq, sa_wk,
                                                             sa_wv, wsaqkv);
    pack_qkv_kernel<<<(3 * E_ * E_) / 256, 256, 0, stream>>>(ca_wq, ca_wk,
                                                             ca_wv, wcaqkv);
    pack_t_kernel<<<(E_ * E_) / 256, 256, 0, stream>>>(sa_wo, wsao, E_, E_);
    pack_t_kernel<<<(E_ * E_) / 256, 256, 0, stream>>>(ca_wo, wcao, E_, E_);
    pack_t_kernel<<<(E_ * 4 * E_) / 256, 256, 0, stream>>>(f_w1, wf1, E_, 4 * E_);
    pack_t_kernel<<<(E_ * 4 * E_) / 256, 256, 0, stream>>>(f_w2, wf2, 4 * E_, E_);
    f32_to_bf16_kernel<<<(M_ * E_) / 256, 256, 0, stream>>>(memory, memb, M_ * E_);

    dim3 ga(B_ * H_, (T_ + 15) / 16);   // 512 x 13

    // ---- self-attention block ----
    ln_kernel<<<M_, 256, 0, stream>>>(idx, ln1_g, ln1_b, hb);
    gemm(hb, wsaqkv, qkvf, nullptr, nullptr, nullptr, 3 * E_, E_, 3 * E_, 0);
    split_qkv_kernel<<<(M_ * 1536) / 256, 256, 0, stream>>>(qkvf, pred_mask,
                                                            qb, kb, vtb, 0);
    attn_kernel<<<ga, 32, 0, stream>>>(qb, kb, vtb, ob);
    gemm(ob, wsao, x1f, nullptr, sa_bo, idx, E_, E_, E_, 1);

    // ---- cross-attention block (k from memory, q & v from x) ----
    ln_kernel<<<M_, 256, 0, stream>>>(x1f, ln2_g, ln2_b, hb);
    gemm(hb,   wcaqkv,                       qkvf,          nullptr, nullptr, nullptr, E_, E_, 3 * E_, 0); // q
    gemm(memb, wcaqkv + (size_t)E_ * E_,     qkvf + E_,     nullptr, nullptr, nullptr, E_, E_, 3 * E_, 0); // k
    gemm(hb,   wcaqkv + (size_t)2 * E_ * E_, qkvf + 2 * E_, nullptr, nullptr, nullptr, E_, E_, 3 * E_, 0); // v
    split_qkv_kernel<<<(M_ * 1536) / 256, 256, 0, stream>>>(qkvf, src_mask,
                                                            qb, kb, vtb, 1);
    attn_kernel<<<ga, 32, 0, stream>>>(qb, kb, vtb, ob);
    gemm(ob, wcao, x2f, nullptr, ca_bo, x1f, E_, E_, E_, 1);

    // ---- feed-forward block ----
    ln_kernel<<<M_, 256, 0, stream>>>(x2f, ln3_g, ln3_b, hb);
    gemm(hb, wf1, nullptr, midb, f_b1, nullptr, 4 * E_, E_, 4 * E_, 2);      // relu
    gemm(midb, wf2, out, nullptr, f_b2, x2f, E_, 4 * E_, E_, 1);             // + resid
}
